// HypergraphLayer_11269994184773
// MI455X (gfx1250) — compile-verified, matrix operand order checked by reading, not run on hardware
//
#include <hip/hip_runtime.h>
#include <hip/hip_bf16.h>
#include <math.h>

#define N_ 32
#define L_ 512
#define P_ 256
#define H_ 8
#define E_ 128
#define DM_ 1024

typedef __bf16 bf16;
typedef __attribute__((ext_vector_type(16))) __bf16 v16bf;
typedef __attribute__((ext_vector_type(8)))  __bf16 v8bf;
typedef __attribute__((ext_vector_type(4)))  __bf16 v4bf;
typedef __attribute__((ext_vector_type(8)))  float  v8f;
typedef __attribute__((ext_vector_type(4)))  float  v4f;

// Fragment storage: [frag][32 lanes][24 elems] (first 16 used).
// Per-lane stride 24 bf16 = 48 B = 12 banks; 16 lanes -> start banks
// {0,4,...,60}: conflict-free ds_load_b128 pairs.
#define FR_STRIDE 24
#define FR_SIZE (32 * FR_STRIDE)

__device__ __forceinline__ float gelu_exact(float v) {
  return 0.5f * v * (1.0f + erff(v * 0.7071067811865475f));
}

// Forward swizzle: element (m,k) of a 16x32 A-frag -> frag-local index.
// An aligned 4-group (or 8-group) of k maps to consecutive j in one lane.
__device__ __forceinline__ int fr_idx_A(int m, int k) {
  int half = (k >> 3) & 1;
  int lane = m | (half << 4);
  int v = ((k & 16) >> 2) + ((k & 7) >> 1);
  return lane * FR_STRIDE + 2 * v + (k & 1);
}
// element (k,n) of a 32x16 B-frag; consecutive n -> consecutive lanes.
__device__ __forceinline__ int fr_idx_B(int k, int n) {
  int half = (k >> 3) & 1;
  int lane = n | (half << 4);
  int v = ((k & 16) >> 2) + ((k & 7) >> 1);
  return lane * FR_STRIDE + 2 * v + (k & 1);
}

__device__ __forceinline__ v16bf load_frag(const bf16* fr, int lane) {
  const v8bf* p = (const v8bf*)(fr + lane * FR_STRIDE);
  v8bf lo = p[0];
  v8bf hi = p[1];
  return __builtin_shufflevector(lo, hi, 0, 1, 2, 3, 4, 5, 6, 7,
                                 8, 9, 10, 11, 12, 13, 14, 15);
}

#define WMMA_BF16(A, B, C) \
  __builtin_amdgcn_wmma_f32_16x16x32_bf16(false, (A), false, (B), (short)0, (C), false, false)

#define MMA_STEP()                                                        \
  do {                                                                    \
    v16bf a0 = load_frag(&AsF[(wm * 2 + 0) * FR_SIZE], lane);             \
    v16bf a1 = load_frag(&AsF[(wm * 2 + 1) * FR_SIZE], lane);             \
    v16bf b0 = load_frag(&BsF[(wn * 4 + 0) * FR_SIZE], lane);             \
    v16bf b1 = load_frag(&BsF[(wn * 4 + 1) * FR_SIZE], lane);             \
    v16bf b2 = load_frag(&BsF[(wn * 4 + 2) * FR_SIZE], lane);             \
    v16bf b3 = load_frag(&BsF[(wn * 4 + 3) * FR_SIZE], lane);             \
    acc[0][0] = WMMA_BF16(a0, b0, acc[0][0]);                             \
    acc[0][1] = WMMA_BF16(a0, b1, acc[0][1]);                             \
    acc[0][2] = WMMA_BF16(a0, b2, acc[0][2]);                             \
    acc[0][3] = WMMA_BF16(a0, b3, acc[0][3]);                             \
    acc[1][0] = WMMA_BF16(a1, b0, acc[1][0]);                             \
    acc[1][1] = WMMA_BF16(a1, b1, acc[1][1]);                             \
    acc[1][2] = WMMA_BF16(a1, b2, acc[1][2]);                             \
    acc[1][3] = WMMA_BF16(a1, b3, acc[1][3]);                             \
  } while (0)

// stage a float4 (along N, consecutive lanes) into B-frag layout
__device__ __forceinline__ void stage_B_n4(bf16* BsF, int k, int c4, v4f w) {
  int idx0 = (c4 >> 4) * FR_SIZE + fr_idx_B(k, c4 & 15);
  BsF[idx0]                 = (bf16)w[0];
  BsF[idx0 + FR_STRIDE]     = (bf16)w[1];
  BsF[idx0 + 2 * FR_STRIDE] = (bf16)w[2];
  BsF[idx0 + 3 * FR_STRIDE] = (bf16)w[3];
}

// -------------------- K1: xh = x_head @ W1 + b1; xv, a_node --------------------
__global__ __launch_bounds__(256) void k1_linear1(
    const float* __restrict__ x, const float* __restrict__ W1,
    const float* __restrict__ b1, const float* __restrict__ W2,
    const float* __restrict__ W4, bf16* __restrict__ xh,
    float* __restrict__ xv, float* __restrict__ anode) {
  __shared__ bf16 AsF[8 * FR_SIZE];
  __shared__ bf16 BsF[8 * FR_SIZE];
  __shared__ float red2[128];
  __shared__ float red4[128];
  int blk = blockIdx.x;
  int lt = blk & 3;
  int nh = blk >> 2;
  int h = nh & (H_ - 1);
  int n = nh >> 3;
  int t = threadIdx.x;
  int lane = t & 31, wid = t >> 5, wm = wid & 3, wn = wid >> 2;
  int half = lane >> 4, nl = lane & 15;

  v8f acc[2][4] = {};
  if (t < 128) { red2[t] = 0.f; red4[t] = 0.f; }

  for (int kk = 0; kk < 4; ++kk) {
#pragma unroll
    for (int i = 0; i < 4; ++i) {      // A: x, contiguous along k -> packed b64
      int e = t + 256 * i;
      int r = e >> 3, c4 = (e & 7) << 2;
      v4f xv4 = *(const v4f*)&x[(size_t)(n * L_ + lt * 128 + r) * DM_ +
                                h * E_ + kk * 32 + c4];
      int idx = (r >> 4) * FR_SIZE + fr_idx_A(r & 15, c4);
      v4bf o = {(bf16)xv4[0], (bf16)xv4[1], (bf16)xv4[2], (bf16)xv4[3]};
      *(v4bf*)&AsF[idx] = o;
    }
#pragma unroll
    for (int i = 0; i < 4; ++i) {      // B: W1, contiguous along n
      int e = t + 256 * i;
      int r = e >> 5, c4 = (e & 31) << 2;
      v4f wv = *(const v4f*)&W1[(kk * 32 + r) * E_ + c4];
      stage_B_n4(BsF, r, c4, wv);
    }
    __syncthreads();
    MMA_STEP();
    __syncthreads();
  }
  size_t base = ((size_t)nh * L_ + lt * 128) * E_;
#pragma unroll
  for (int mi = 0; mi < 2; ++mi) {
#pragma unroll
    for (int r = 0; r < 8; ++r) {
      int rowb = wm * 32 + mi * 16 + 8 * half + r;
      float p2 = 0.f, p4 = 0.f;
#pragma unroll
      for (int ni = 0; ni < 4; ++ni) {
        int col = wn * 64 + ni * 16 + nl;
        float v = acc[mi][ni][r] + b1[col];
        xh[base + (size_t)rowb * E_ + col] = (bf16)v;
        float g = gelu_exact(v);
        p2 += g * W2[col];
        p4 += g * W4[E_ + col];
      }
#pragma unroll
      for (int off = 1; off < 16; off <<= 1) {
        p2 += __shfl_xor(p2, off, 16);
        p4 += __shfl_xor(p4, off, 16);
      }
      if (nl == 0) { atomicAdd(&red2[rowb], p2); atomicAdd(&red4[rowb], p4); }
    }
  }
  __syncthreads();
  if (t < 128) {
    int row = lt * 128 + t;
    xv[(size_t)nh * L_ + row] = red2[t];     // b2 constant over heads: cancels
    anode[(size_t)nh * L_ + row] = red4[t];  // b4 cancels in softmax
  }
}

// -------------------- K2a: w = softmax over heads of xv --------------------
__global__ __launch_bounds__(256) void k2_softmax_w(
    const float* __restrict__ xv, float* __restrict__ w) {
  int idx = blockIdx.x * 256 + threadIdx.x;  // n*L + l
  if (idx >= N_ * L_) return;
  int n = idx / L_, l = idx - n * L_;
  float v[H_]; float mx = -3.0e38f;
#pragma unroll
  for (int h = 0; h < H_; ++h) {
    v[h] = xv[(size_t)(n * H_ + h) * L_ + l];
    mx = fmaxf(mx, v[h]);
  }
  float s = 0.f;
#pragma unroll
  for (int h = 0; h < H_; ++h) { v[h] = __expf(v[h] - mx); s += v[h]; }
  float inv = 1.f / s;
#pragma unroll
  for (int h = 0; h < H_; ++h) w[(size_t)(n * H_ + h) * L_ + l] = v[h] * inv;
}

// -------------------- K2b: colsum over L of xh --------------------
__global__ __launch_bounds__(128) void k2b_colsum(
    const bf16* __restrict__ xh, float* __restrict__ cs) {
  int nh = blockIdx.x;
  int e = threadIdx.x;
  const bf16* p = xh + (size_t)nh * L_ * E_ + e;
  float s = 0.f;
  for (int l = 0; l < L_; ++l) s += (float)p[(size_t)l * E_];
  cs[nh * E_ + e] = s;
}

// ---------- K3: t = gelu( AdjT @ ((w-1/8) xh) + (1/8) colsum ) ----------
__global__ __launch_bounds__(256) void k3_edge_agg(
    const float* __restrict__ adj, const bf16* __restrict__ xh,
    const float* __restrict__ w, const float* __restrict__ cs,
    bf16* __restrict__ tmat) {
  __shared__ bf16 AsF[8 * FR_SIZE];
  __shared__ bf16 BsF[8 * FR_SIZE];
  int blk = blockIdx.x;
  int pt = blk & 1;
  int nh = blk >> 1;
  int n = nh >> 3;
  int t = threadIdx.x;
  int lane = t & 31, wid = t >> 5, wm = wid & 3, wn = wid >> 2;
  int half = lane >> 4, nl = lane & 15;

  v8f acc[2][4] = {};
  for (int kk = 0; kk < 16; ++kk) {      // K = L = 512
#pragma unroll
    for (int i = 0; i < 4; ++i) {        // A: adj, contiguous along p (= M rows)
      int e = t + 256 * i;
      int c = e >> 5, r4 = (e & 31) << 2;
      v4f av = *(const v4f*)&adj[(size_t)(n * L_ + kk * 32 + c) * P_ +
                                 pt * 128 + r4];
      int idx0 = (r4 >> 4) * FR_SIZE + fr_idx_A(r4 & 15, c);
      AsF[idx0]                 = (bf16)av[0];   // consecutive m -> +FR_STRIDE
      AsF[idx0 + FR_STRIDE]     = (bf16)av[1];
      AsF[idx0 + 2 * FR_STRIDE] = (bf16)av[2];
      AsF[idx0 + 3 * FR_STRIDE] = (bf16)av[3];
    }
#pragma unroll
    for (int i = 0; i < 2; ++i) {        // B: (w-1/8)*xh, contiguous along e
      int e = t + 256 * i;
      int r = e >> 4, c8 = (e & 15) << 3;
      int l = kk * 32 + r;
      float ww = w[(size_t)nh * L_ + l] - 0.125f;
      v8bf hv = *(const v8bf*)&xh[((size_t)nh * L_ + l) * E_ + c8];
      int idx0 = (c8 >> 4) * FR_SIZE + fr_idx_B(r, c8 & 15);
#pragma unroll
      for (int q = 0; q < 8; ++q)
        BsF[idx0 + q * FR_STRIDE] = (bf16)(ww * (float)hv[q]);
    }
    __syncthreads();
    MMA_STEP();
    __syncthreads();
  }
#pragma unroll
  for (int mi = 0; mi < 2; ++mi) {
#pragma unroll
    for (int r = 0; r < 8; ++r) {
      int rowb = wm * 32 + mi * 16 + 8 * half + r;
#pragma unroll
      for (int ni = 0; ni < 4; ++ni) {
        int col = wn * 64 + ni * 16 + nl;
        float v = acc[mi][ni][r] + 0.125f * cs[nh * E_ + col];
        tmat[((size_t)nh * P_ + pt * 128 + rowb) * E_ + col] = (bf16)gelu_exact(v);
      }
    }
  }
}

// -------------------- K4: he = t @ W3 + b3; a_edge --------------------
__global__ __launch_bounds__(256) void k4_linear3(
    const bf16* __restrict__ tmat, const float* __restrict__ W3,
    const float* __restrict__ b3, const float* __restrict__ W4,
    bf16* __restrict__ he, float* __restrict__ aedge) {
  __shared__ bf16 AsF[8 * FR_SIZE];
  __shared__ bf16 BsF[8 * FR_SIZE];
  __shared__ float red1[128];
  int blk = blockIdx.x;
  int pt = blk & 1;
  int nh = blk >> 1;
  int t = threadIdx.x;
  int lane = t & 31, wid = t >> 5, wm = wid & 3, wn = wid >> 2;
  int half = lane >> 4, nl = lane & 15;

  v8f acc[2][4] = {};
  if (t < 128) red1[t] = 0.f;
  for (int kk = 0; kk < 4; ++kk) {       // K = E = 128
#pragma unroll
    for (int i = 0; i < 2; ++i) {        // A: tmat bf16, aligned 8 along k ->
      int e = t + 256 * i;               //    j0..j0+7 in one lane: b128 copy
      int r = e >> 2, k8 = (e & 3) << 3;
      v8bf tv = *(const v8bf*)&tmat[((size_t)nh * P_ + pt * 128 + r) * E_ +
                                    kk * 32 + k8];
      int idx = (r >> 4) * FR_SIZE + fr_idx_A(r & 15, k8);
      *(v8bf*)&AsF[idx] = tv;
    }
#pragma unroll
    for (int i = 0; i < 4; ++i) {        // B: W3, contiguous along n
      int e = t + 256 * i;
      int r = e >> 5, c4 = (e & 31) << 2;
      v4f wv = *(const v4f*)&W3[(kk * 32 + r) * E_ + c4];
      stage_B_n4(BsF, r, c4, wv);
    }
    __syncthreads();
    MMA_STEP();
    __syncthreads();
  }
  size_t base = ((size_t)nh * P_ + pt * 128) * E_;
#pragma unroll
  for (int mi = 0; mi < 2; ++mi) {
#pragma unroll
    for (int r = 0; r < 8; ++r) {
      int rowb = wm * 32 + mi * 16 + 8 * half + r;
      float p1 = 0.f;
#pragma unroll
      for (int ni = 0; ni < 4; ++ni) {
        int col = wn * 64 + ni * 16 + nl;
        float v = acc[mi][ni][r] + b3[col];
        he[base + (size_t)rowb * E_ + col] = (bf16)v;
        p1 += gelu_exact(v) * W4[col];
      }
#pragma unroll
      for (int off = 1; off < 16; off <<= 1) p1 += __shfl_xor(p1, off, 16);
      if (nl == 0) atomicAdd(&red1[rowb], p1);
    }
  }
  __syncthreads();
  if (t < 128) aedge[(size_t)nh * P_ + pt * 128 + t] = red1[t];
}

// ------- K5: per (n,l): head-softmax max/denominator over p; no_edge -------
__global__ __launch_bounds__(256) void k5_node_stats(
    const float* __restrict__ adj, const float* __restrict__ anode,
    const float* __restrict__ aedge, float* __restrict__ mv,
    float* __restrict__ dv, float* __restrict__ noedge) {
  __shared__ float an_s[H_];
  __shared__ float red[256];
  int idx = blockIdx.x;              // n*L + l
  int n = idx / L_, l = idx - n * L_;
  int p = threadIdx.x;
  if (p < H_) an_s[p] = anode[(size_t)(n * H_ + p) * L_ + l];
  __syncthreads();
  float s[H_];
  float mx = -3.0e38f;
#pragma unroll
  for (int h = 0; h < H_; ++h) {
    s[h] = an_s[h] + aedge[(size_t)(n * H_ + h) * P_ + p];
    mx = fmaxf(mx, s[h]);
  }
  float d = 0.f;
#pragma unroll
  for (int h = 0; h < H_; ++h) d += __expf(s[h] - mx);
  mv[(size_t)idx * P_ + p] = mx;
  dv[(size_t)idx * P_ + p] = d;
  red[p] = adj[(size_t)idx * P_ + p];
  __syncthreads();
  for (int off = 128; off > 0; off >>= 1) {
    if (p < off) red[p] += red[p + off];
    __syncthreads();
  }
  if (p == 0) noedge[idx] = (red[0] == 0.f) ? 1.f : 0.f;
}

// -------------- K6: out = gelu( Att @ he ), with passthrough --------------
__global__ __launch_bounds__(256) void k6_node_out(
    const float* __restrict__ adj, const bf16* __restrict__ he,
    const float* __restrict__ anode, const float* __restrict__ aedge,
    const float* __restrict__ mv, const float* __restrict__ dv,
    const float* __restrict__ noedge, const float* __restrict__ x,
    float* __restrict__ out) {
  __shared__ bf16 AsF[8 * FR_SIZE];
  __shared__ bf16 BsF[8 * FR_SIZE];
  __shared__ float an_s[128];
  int blk = blockIdx.x;
  int lt = blk & 3;
  int nh = blk >> 2;
  int h = nh & (H_ - 1);
  int n = nh >> 3;
  int t = threadIdx.x;
  int lane = t & 31, wid = t >> 5, wm = wid & 3, wn = wid >> 2;
  int half = lane >> 4, nl = lane & 15;

  if (t < 128) an_s[t] = anode[(size_t)nh * L_ + lt * 128 + t];
  __syncthreads();

  v8f acc[2][4] = {};
  for (int kk = 0; kk < 8; ++kk) {        // K = P = 256
#pragma unroll
    for (int i = 0; i < 4; ++i) {         // A: att[l, p]; p contiguous along k
      int e = t + 256 * i;
      int r = e >> 3, c4 = (e & 7) << 2;
      int l = lt * 128 + r;
      int p0 = kk * 32 + c4;
      size_t ai = (size_t)(n * L_ + l) * P_ + p0;
      v4f a4 = *(const v4f*)&adj[ai];
      v4f m4 = *(const v4f*)&mv[ai];
      v4f d4 = *(const v4f*)&dv[ai];
      v4f e4 = *(const v4f*)&aedge[(size_t)nh * P_ + p0];
      float an_r = an_s[r];
      v4bf o;
#pragma unroll
      for (int q = 0; q < 4; ++q) {
        float aval = (a4[q] != 0.f) ? __expf(an_r + e4[q] - m4[q]) / d4[q]
                                    : 0.125f;   // all-masked softmax = 1/8
        o[q] = (bf16)aval;
      }
      int idx = (r >> 4) * FR_SIZE + fr_idx_A(r & 15, c4);
      *(v4bf*)&AsF[idx] = o;
    }
#pragma unroll
    for (int i = 0; i < 2; ++i) {         // B: he bf16, contiguous along e
      int e = t + 256 * i;
      int r = e >> 4, c8 = (e & 15) << 3;
      v8bf hv = *(const v8bf*)&he[((size_t)nh * P_ + kk * 32 + r) * E_ + c8];
      int idx0 = (c8 >> 4) * FR_SIZE + fr_idx_B(r, c8 & 15);
#pragma unroll
      for (int q = 0; q < 8; ++q) BsF[idx0 + q * FR_STRIDE] = hv[q];
    }
    __syncthreads();
    MMA_STEP();
    __syncthreads();
  }
#pragma unroll
  for (int mi = 0; mi < 2; ++mi) {
#pragma unroll
    for (int r = 0; r < 8; ++r) {
      int row = lt * 128 + wm * 32 + mi * 16 + 8 * half + r;
      float ne = noedge[n * L_ + row];
#pragma unroll
      for (int ni = 0; ni < 4; ++ni) {
        int col = wn * 64 + ni * 16 + nl;
        size_t oi = (size_t)(n * L_ + row) * DM_ + h * E_ + col;
        float g = gelu_exact(acc[mi][ni][r]);
        out[oi] = (ne != 0.f) ? x[oi] : g;
      }
    }
  }
}

// -------------------- workspace layout (bytes) --------------------
static const size_t OFF_XH = 0;                       // bf16 N*H*L*E  = 32 MiB
static const size_t OFF_T  = 33554432;                // bf16 N*H*P*E  = 16 MiB
static const size_t OFF_HE = 50331648;                // bf16 N*H*P*E  = 16 MiB
static const size_t OFF_XV = 67108864;                // f32  N*H*L
static const size_t OFF_AN = 67633152;                // f32  N*H*L
static const size_t OFF_W  = 68157440;                // f32  N*H*L
static const size_t OFF_AE = 68681728;                // f32  N*H*P
static const size_t OFF_CS = 68943872;                // f32  N*H*E
static const size_t OFF_M  = 69074944;                // f32  N*L*P    = 16 MiB
static const size_t OFF_D  = 85852160;                // f32  N*L*P    = 16 MiB
static const size_t OFF_NE = 102629376;               // f32  N*L

extern "C" void kernel_launch(void* const* d_in, const int* in_sizes, int n_in,
                              void* d_out, int out_size, void* d_ws, size_t ws_size,
                              hipStream_t stream) {
  (void)in_sizes; (void)n_in; (void)out_size; (void)ws_size;
  const float* x   = (const float*)d_in[0];
  const float* adj = (const float*)d_in[1];
  const float* W1  = (const float*)d_in[2];
  const float* b1  = (const float*)d_in[3];
  const float* W2  = (const float*)d_in[4];
  // b2 (d_in[5]) cancels in the head softmax
  const float* W3  = (const float*)d_in[6];
  const float* b3  = (const float*)d_in[7];
  const float* W4  = (const float*)d_in[8];
  // b4 (d_in[9]) cancels in the head softmax
  float* out = (float*)d_out;
  char* ws = (char*)d_ws;

  bf16*  xh    = (bf16*)(ws + OFF_XH);
  bf16*  tmat  = (bf16*)(ws + OFF_T);
  bf16*  he    = (bf16*)(ws + OFF_HE);
  float* xv    = (float*)(ws + OFF_XV);
  float* an    = (float*)(ws + OFF_AN);
  float* w     = (float*)(ws + OFF_W);
  float* ae    = (float*)(ws + OFF_AE);
  float* cs    = (float*)(ws + OFF_CS);
  float* mv    = (float*)(ws + OFF_M);
  float* dv    = (float*)(ws + OFF_D);
  float* ne    = (float*)(ws + OFF_NE);

  k1_linear1<<<N_ * H_ * (L_ / 128), 256, 0, stream>>>(x, W1, b1, W2, W4, xh, xv, an);
  k2_softmax_w<<<(N_ * L_) / 256, 256, 0, stream>>>(xv, w);
  k2b_colsum<<<N_ * H_, 128, 0, stream>>>(xh, cs);
  k3_edge_agg<<<N_ * H_ * (P_ / 128), 256, 0, stream>>>(adj, xh, w, cs, tmat);
  k4_linear3<<<N_ * H_ * (P_ / 128), 256, 0, stream>>>(tmat, W3, b3, W4, he, ae);
  k5_node_stats<<<N_ * L_, 256, 0, stream>>>(adj, an, ae, mv, dv, ne);
  k6_node_out<<<N_ * H_ * (L_ / 128), 256, 0, stream>>>(adj, he, an, ae, mv, dv, ne, x, out);
}
